// GATLayer_41205916238261
// MI455X (gfx1250) — compile-verified, matrix operand order checked by reading, not run on hardware
//
#include <hip/hip_runtime.h>

#define NU 100000
#define NI 50000
#define NE 800000
#define DIN 256
#define DH 128
#define NH 4
#define SLOPE 0.2f

typedef __attribute__((ext_vector_type(16))) __bf16 v16bf;
typedef __attribute__((ext_vector_type(8)))  float  v8f;

__device__ __forceinline__ float lrelu(float x) { return x > 0.f ? x : SLOPE * x; }

// monotone float <-> ordered-uint mapping so atomicMax(unsigned) == float max
__device__ __forceinline__ unsigned f2ord(float f) {
  unsigned u = __float_as_uint(f);
  return (u & 0x80000000u) ? ~u : (u | 0x80000000u);
}
__device__ __forceinline__ float ord2f(unsigned u) {
  return (u & 0x80000000u) ? __uint_as_float(u ^ 0x80000000u)
                           : __uint_as_float(~u);
}

// ---------------------------------------------------------------------------
// Pre-pack W[DIN, DH] (f32, row-major) into bf16 WMMA B-fragments:
// P[((kstep*8 + tile)*32 + lane)*16 + e]  — one contiguous 32B vector per lane.
// grid = 64 blocks (kstep*8+tile), block = 32 (lane)
// ---------------------------------------------------------------------------
__global__ __launch_bounds__(32)
void pack_w_kernel(const float* __restrict__ W, __bf16* __restrict__ P)
{
  const int lane  = threadIdx.x & 31;
  const int kstep = blockIdx.x >> 3;   // 0..7  (K block of 32)
  const int t     = blockIdx.x & 7;    // 0..7  (N tile of 16)
  const int half  = lane >> 4;
  const int mr    = lane & 15;
  const int col   = t * 16 + mr;
  const int kb    = kstep * 32 + half * 8;
  __bf16* dst = P + ((size_t)(kstep * 8 + t) * 32 + lane) * 16;
#pragma unroll
  for (int r = 0; r < 8; ++r) {
    dst[r]     = (__bf16)W[(size_t)(kb + r) * DH + col];
    dst[8 + r] = (__bf16)W[(size_t)(kb + 16 + r) * DH + col];
  }
}

// ---------------------------------------------------------------------------
// out[M, DH] = A[M, DIN] @ W + bias.  One wave computes a 16x128 strip:
// A fragment loaded once per K-step, 8 WMMAs against pre-packed B fragments.
// grid = M/16, block = 32
// ---------------------------------------------------------------------------
__global__ __launch_bounds__(32)
void proj_gemm_kernel(const float* __restrict__ A, const __bf16* __restrict__ WP,
                      const float* __restrict__ bias, float* __restrict__ out, int M)
{
  const int lane = threadIdx.x & 31;
  const int half = lane >> 4;            // K-half selector within fragment
  const int mr   = lane & 15;
  const int rowA = blockIdx.x * 16 + mr; // A-matrix row this lane feeds
  const float* __restrict__ arow = A + (size_t)(rowA < M ? rowA : M - 1) * DIN;

  v8f acc[8] = {};
#pragma unroll 2
  for (int ks = 0; ks < 8; ++ks) {
    // A fragment (16-bit A 16x32 layout): elem e<8 -> K = ks*32 + half*8 + e
    //                                     elem e>=8 -> K = ks*32 + 16 + half*8 + (e-8)
    const float* ap = arow + ks * 32 + half * 8;
    __builtin_prefetch(ap + 32, 0, 3);   // WGP-scope prefetch of next K block
    v16bf af;
#pragma unroll
    for (int r = 0; r < 8; ++r) {
      af[r]     = (__bf16)ap[r];
      af[8 + r] = (__bf16)ap[16 + r];
    }
    const v16bf* wp = (const v16bf*)WP + (size_t)ks * 8 * 32 + lane;
#pragma unroll
    for (int t = 0; t < 8; ++t) {
      v16bf bf = wp[t * 32];             // contiguous 32B fragment load
      acc[t] = __builtin_amdgcn_wmma_f32_16x16x32_bf16(
          /*neg_a=*/false, af, /*neg_b=*/false, bf,
          /*c_mod=*/(short)0, acc[t], /*reuse_a=*/false, /*reuse_b=*/false);
    }
  }
  // C layout: lane n = lane%16 is the column; rows = half*8 + v
#pragma unroll
  for (int t = 0; t < 8; ++t) {
    const int col = t * 16 + mr;
    const float bcol = bias[col];
#pragma unroll
    for (int v = 0; v < 8; ++v) {
      int row = blockIdx.x * 16 + half * 8 + v;
      if (row < M) out[(size_t)row * DH + col] = acc[t][v] + bcol;
    }
  }
}

// ---------------------------------------------------------------------------
// s[n, 4] = h[n, :] . a[h, :]  for 4 heads (a is 4x128, hot in cache)
// ---------------------------------------------------------------------------
__global__ __launch_bounds__(256)
void scores_kernel(const float* __restrict__ h, const float* __restrict__ a,
                   float* __restrict__ s, int N)
{
  int n = blockIdx.x * blockDim.x + threadIdx.x;
  if (n >= N) return;
  const float4* hr = (const float4*)(h + (size_t)n * DH);
  const float4* av = (const float4*)a;
  float s0 = 0.f, s1 = 0.f, s2 = 0.f, s3 = 0.f;
#pragma unroll 8
  for (int i = 0; i < DH / 4; ++i) {
    float4 hv = hr[i];
    float4 a0 = av[i], a1 = av[32 + i], a2 = av[64 + i], a3 = av[96 + i];
    s0 += hv.x * a0.x + hv.y * a0.y + hv.z * a0.z + hv.w * a0.w;
    s1 += hv.x * a1.x + hv.y * a1.y + hv.z * a1.z + hv.w * a1.w;
    s2 += hv.x * a2.x + hv.y * a2.y + hv.z * a2.z + hv.w * a2.w;
    s3 += hv.x * a3.x + hv.y * a3.y + hv.z * a3.z + hv.w * a3.w;
  }
  ((float4*)s)[n] = make_float4(s0, s1, s2, s3);
}

__global__ __launch_bounds__(256)
void fill_kernel(unsigned* __restrict__ p, unsigned v, int n)
{
  int i = blockIdx.x * blockDim.x + threadIdx.x;
  if (i < n) p[i] = v;
}

// segment max of leaky_relu(s_src[src]+s_dst[dst]) into m_ord (ordered uints)
__global__ __launch_bounds__(256)
void edge_max_kernel(const float* __restrict__ s_src, const float* __restrict__ s_dst,
                     const int* __restrict__ src, const int* __restrict__ dst,
                     unsigned* __restrict__ m_ord, int E)
{
  int e = blockIdx.x * blockDim.x + threadIdx.x;
  if (e >= E) return;
  int si = src[e], di = dst[e];
  float4 a = ((const float4*)s_src)[si];
  float4 b = ((const float4*)s_dst)[di];
  unsigned* m = m_ord + (size_t)di * NH;
  atomicMax(m + 0, f2ord(lrelu(a.x + b.x)));
  atomicMax(m + 1, f2ord(lrelu(a.y + b.y)));
  atomicMax(m + 2, f2ord(lrelu(a.z + b.z)));
  atomicMax(m + 3, f2ord(lrelu(a.w + b.w)));
}

// e = exp(logit - m[dst]); store (unnormalized) att, accumulate denom
__global__ __launch_bounds__(256)
void edge_expsum_kernel(const float* __restrict__ s_src, const float* __restrict__ s_dst,
                        const int* __restrict__ src, const int* __restrict__ dst,
                        const unsigned* __restrict__ m_ord, float* __restrict__ denom,
                        float* __restrict__ att, int E)
{
  int e = blockIdx.x * blockDim.x + threadIdx.x;
  if (e >= E) return;
  int si = src[e], di = dst[e];
  float4 a = ((const float4*)s_src)[si];
  float4 b = ((const float4*)s_dst)[di];
  uint4  m = ((const uint4*)m_ord)[di];
  float e0 = __expf(lrelu(a.x + b.x) - ord2f(m.x));
  float e1 = __expf(lrelu(a.y + b.y) - ord2f(m.y));
  float e2 = __expf(lrelu(a.z + b.z) - ord2f(m.z));
  float e3 = __expf(lrelu(a.w + b.w) - ord2f(m.w));
  ((float4*)att)[e] = make_float4(e0, e1, e2, e3);
  float* d = denom + (size_t)di * NH;
  atomicAdd(d + 0, e0);
  atomicAdd(d + 1, e1);
  atomicAdd(d + 2, e2);
  atomicAdd(d + 3, e3);
}

__global__ __launch_bounds__(256)
void att_norm_kernel(float* __restrict__ att, const float* __restrict__ denom,
                     const int* __restrict__ dst, int E)
{
  int e = blockIdx.x * blockDim.x + threadIdx.x;
  if (e >= E) return;
  float4 at = ((float4*)att)[e];
  float4 dn = ((const float4*)denom)[dst[e]];
  at.x /= dn.x; at.y /= dn.y; at.z /= dn.z; at.w /= dn.w;
  ((float4*)att)[e] = at;
}

// z[dst, :] += att[e,h] * h_src[src, :]   (wave = one edge; lane = float4 chunk)
__global__ __launch_bounds__(256)
void scatter_kernel(const float* __restrict__ att, int h,
                    const float* __restrict__ hsrc,
                    const int* __restrict__ src, const int* __restrict__ dst,
                    float* __restrict__ z, int E)
{
  int idx = blockIdx.x * blockDim.x + threadIdx.x;
  int e = idx >> 5;          // 32 chunks of 4 floats = 128 features
  int c = idx & 31;
  if (e >= E) return;
  float a = att[(size_t)e * NH + h];
  int si = src[e], di = dst[e];
  float4 v = ((const float4*)(hsrc + (size_t)si * DH))[c];
  float* zp = z + (size_t)di * DH + c * 4;
  atomicAdd(zp + 0, a * v.x);
  atomicAdd(zp + 1, a * v.y);
  atomicAdd(zp + 2, a * v.z);
  atomicAdd(zp + 3, a * v.w);
}

// out += elu(z) / NH
__global__ __launch_bounds__(256)
void elu_merge_kernel(const float* __restrict__ z, float* __restrict__ out, int total)
{
  int i = blockIdx.x * blockDim.x + threadIdx.x;
  if (i >= total) return;
  float v = z[i];
  float r = v > 0.f ? v : (__expf(v) - 1.f);
  out[i] += r * (1.f / NH);
}

// ---------------------------------------------------------------------------
static void run_gat(const float* hsrc, int Nsrc, const float* hdst, int Ndst,
                    const float* a_src, const float* a_dst,
                    const int* src, const int* dst, int E,
                    float* s_src, float* s_dst, unsigned* m_ord, float* denom,
                    float* att, float* z, float* out, hipStream_t stream)
{
  const int T = 256;
  scores_kernel<<<(Nsrc + T - 1) / T, T, 0, stream>>>(hsrc, a_src, s_src, Nsrc);
  scores_kernel<<<(Ndst + T - 1) / T, T, 0, stream>>>(hdst, a_dst, s_dst, Ndst);
  int nh = Ndst * NH;
  fill_kernel<<<(nh + T - 1) / T, T, 0, stream>>>(m_ord, 0u, nh);       // == -inf side
  fill_kernel<<<(nh + T - 1) / T, T, 0, stream>>>((unsigned*)denom, 0u, nh);
  edge_max_kernel<<<(E + T - 1) / T, T, 0, stream>>>(s_src, s_dst, src, dst, m_ord, E);
  edge_expsum_kernel<<<(E + T - 1) / T, T, 0, stream>>>(s_src, s_dst, src, dst,
                                                        m_ord, denom, att, E);
  att_norm_kernel<<<(E + T - 1) / T, T, 0, stream>>>(att, denom, dst, E);

  int tot = Ndst * DH;
  fill_kernel<<<(tot + T - 1) / T, T, 0, stream>>>((unsigned*)out, 0u, tot);
  long long sthreads = (long long)E * 32;
  int sblocks = (int)((sthreads + T - 1) / T);
  for (int h = 0; h < NH; ++h) {
    fill_kernel<<<(tot + T - 1) / T, T, 0, stream>>>((unsigned*)z, 0u, tot);
    scatter_kernel<<<sblocks, T, 0, stream>>>(att, h, hsrc, src, dst, z, E);
    elu_merge_kernel<<<(tot + T - 1) / T, T, 0, stream>>>(z, out, tot);
  }
}

extern "C" void kernel_launch(void* const* d_in, const int* in_sizes, int n_in,
                              void* d_out, int out_size, void* d_ws, size_t ws_size,
                              hipStream_t stream)
{
  const float* h_user     = (const float*)d_in[0];
  const float* h_item     = (const float*)d_in[1];
  const float* w_user     = (const float*)d_in[2];
  const float* b_user     = (const float*)d_in[3];
  const float* w_item     = (const float*)d_in[4];
  const float* b_item     = (const float*)d_in[5];
  const float* a_user_src = (const float*)d_in[6];
  const float* a_user_dst = (const float*)d_in[7];
  const float* a_item_src = (const float*)d_in[8];
  const float* a_item_dst = (const float*)d_in[9];
  const int*   i2u_src    = (const int*)d_in[10];
  const int*   i2u_dst    = (const int*)d_in[11];
  const int*   u2i_src    = (const int*)d_in[12];
  const int*   u2i_dst    = (const int*)d_in[13];

  float* out    = (float*)d_out;
  float* hu_new = out;                          // NU*DH
  float* hi_new = out + (size_t)NU * DH;        // NI*DH

  // workspace layout (floats). z aliases hu: hu is only read at the start of
  // layer 1 (s_dst scores) and is unused afterwards — stream order makes the
  // reuse safe, and every buffer is rewritten each call (deterministic).
  float*    ws    = (float*)d_ws;
  float*    hu    = ws;                          // NU*DH   (later reused as z)
  float*    hi    = hu + (size_t)NU * DH;        // NI*DH
  float*    s_src = hi + (size_t)NI * DH;        // NU*NH
  float*    s_dst = s_src + (size_t)NU * NH;     // NU*NH
  unsigned* m_ord = (unsigned*)(s_dst + (size_t)NU * NH); // NU*NH
  float*    denom = (float*)m_ord + (size_t)NU * NH;      // NU*NH
  float*    att   = denom + (size_t)NU * NH;     // NE*NH
  __bf16*   wpu   = (__bf16*)(att + (size_t)NE * NH);     // DIN*DH bf16 (packed)
  __bf16*   wpi   = wpu + (size_t)DIN * DH;               // DIN*DH bf16 (packed)
  float*    z     = hu;

  // pre-pack both weight matrices into WMMA B-fragment bf16 layout
  pack_w_kernel<<<64, 32, 0, stream>>>(w_user, wpu);
  pack_w_kernel<<<64, 32, 0, stream>>>(w_item, wpi);

  // dense projections via WMMA (NU, NI both divisible by 16)
  proj_gemm_kernel<<<NU / 16, 32, 0, stream>>>(h_user, wpu, b_user, hu, NU);
  proj_gemm_kernel<<<NI / 16, 32, 0, stream>>>(h_item, wpi, b_item, hi, NI);

  // layer 1: items -> users (reads hu only in its first kernel, then z clobbers it)
  run_gat(hi, NI, hu, NU, a_user_src, a_user_dst, i2u_src, i2u_dst, NE,
          s_src, s_dst, m_ord, denom, att, z, hu_new, stream);

  // layer 2: updated users -> items
  run_gat(hu_new, NU, hi, NI, a_item_src, a_item_dst, u2i_src, u2i_dst, NE,
          s_src, s_dst, m_ord, denom, att, z, hi_new, stream);
}
